// CausalMoE_5291399709239
// MI455X (gfx1250) — compile-verified, matrix operand order checked by reading, not run on hardware
//
#include <hip/hip_runtime.h>
#include <cstdint>

typedef __attribute__((ext_vector_type(16))) __bf16 bf16x16;
typedef __attribute__((ext_vector_type(8)))  float  f32x8;

// ---------- helpers ----------
__device__ __forceinline__ unsigned short f2bf(float f) {
    unsigned int u = __float_as_uint(f);
    u += 0x7fffu + ((u >> 16) & 1u);      // round-to-nearest-even
    return (unsigned short)(u >> 16);
}

__device__ __forceinline__ float gelu_exact(float x) {
    return 0.5f * x * (1.0f + erff(x * 0.70710678118654752f));
}

// Async-copy 32 bytes global -> LDS (2 x b128), tracked on ASYNCcnt, no VGPR round-trip.
// INST_OFFSET applies to both the global and LDS side per the ISA pseudocode.
__device__ __forceinline__ void async_copy32(unsigned lds_off, const void* gptr) {
    asm volatile(
        "global_load_async_to_lds_b128 %0, %1, off\n\t"
        "global_load_async_to_lds_b128 %0, %1, off offset:16"
        :: "v"(lds_off), "v"(gptr) : "memory");
}

__device__ __forceinline__ void wait_async0() {
    asm volatile("s_wait_asynccnt 0" ::: "memory");
}

// ---------- kernel 1/2: fp32 [R x C] -> bf16 [C x R] (transpose + convert) ----------
__global__ __launch_bounds__(256) void k_transpose_f32_bf16(
    const float* __restrict__ in, unsigned short* __restrict__ out, int R, int C) {
    __shared__ float tile[32][33];
    const int bx = blockIdx.x * 32;   // C direction
    const int by = blockIdx.y * 32;   // R direction
    const int tx = threadIdx.x, ty = threadIdx.y;   // block (32,8)
#pragma unroll
    for (int j = 0; j < 32; j += 8)
        tile[ty + j][tx] = in[(size_t)(by + ty + j) * C + (bx + tx)];
    __syncthreads();
#pragma unroll
    for (int j = 0; j < 32; j += 8)
        out[(size_t)(bx + ty + j) * R + (by + tx)] = f2bf(tile[tx][ty + j]);
}

// ---------- kernel 3: causal extract + mechanisms + routing; emits bf16 "modified" ----------
// One wave (32 lanes) per token. 8 waves per block -> 512 blocks for 4096 tokens.
__global__ __launch_bounds__(256) void k_causal_modify(
    const float* __restrict__ hsd, const float* __restrict__ Pe,
    const float* __restrict__ Pr,  const float* __restrict__ Adj,
    const float* __restrict__ W1,  const float* __restrict__ b1,
    const float* __restrict__ W2,  const float* __restrict__ b2,
    unsigned short* __restrict__ modified) {
    const int H = 2048;
    const int wave = threadIdx.x >> 5;
    const int lane = threadIdx.x & 31;
    const int token = blockIdx.x * 8 + wave;
    const float* __restrict__ hrow = hsd + (size_t)token * H;

    // cv accumulation: each lane handles strided k, 16 parallel dot products
    float a0=0,a1=0,a2=0,a3=0,a4=0,a5=0,a6=0,a7=0,a8=0,a9=0,a10=0,a11=0,a12=0,a13=0,a14=0,a15=0;
    for (int k = lane; k < H; k += 32) {
        const float x = hrow[k];
        const float4* pe4 = (const float4*)(Pe + k * 16);
        float4 p0 = pe4[0], p1 = pe4[1], p2 = pe4[2], p3 = pe4[3];
        a0  += x * p0.x; a1  += x * p0.y; a2  += x * p0.z; a3  += x * p0.w;
        a4  += x * p1.x; a5  += x * p1.y; a6  += x * p1.z; a7  += x * p1.w;
        a8  += x * p2.x; a9  += x * p2.y; a10 += x * p2.z; a11 += x * p2.w;
        a12 += x * p3.x; a13 += x * p3.y; a14 += x * p3.z; a15 += x * p3.w;
    }
    float accg[16] = {a0,a1,a2,a3,a4,a5,a6,a7,a8,a9,a10,a11,a12,a13,a14,a15};
    float cv[16];
#pragma unroll
    for (int g = 0; g < 16; ++g) {
        float v = accg[g];
        v += __shfl_xor(v, 16, 32);
        v += __shfl_xor(v, 8, 32);
        v += __shfl_xor(v, 4, 32);
        v += __shfl_xor(v, 2, 32);
        v += __shfl_xor(v, 1, 32);
        cv[g] = tanhf(v);        // every lane holds all 16 cv values
    }

    // mechanism m on lane (m = lane&15; lanes 16..31 duplicate 0..15)
    const int m = lane & 15;
    float pi = 0.f;
#pragma unroll
    for (int g = 0; g < 16; ++g) pi += cv[g] * Adj[g * 16 + m];
    float eff = b2[m];
#pragma unroll
    for (int j = 0; j < 8; ++j) {
        float s = b1[m * 8 + j] + pi * W1[(m * 17 + 16) * 8 + j];
#pragma unroll
        for (int v = 0; v < 16; ++v) s += cv[v] * W1[(m * 17 + v) * 8 + j];
        eff += gelu_exact(s) * W2[m * 8 + j];
    }
    float effs[16];
#pragma unroll
    for (int g = 0; g < 16; ++g) effs[g] = __shfl(eff, g, 32);

    // routing: modified = hs + 0.5 * effects @ P_route, stored bf16
    for (int i = lane; i < H; i += 32) {
        float s = 0.f;
#pragma unroll
        for (int g = 0; g < 16; ++g) s += effs[g] * Pr[g * H + i];
        modified[(size_t)token * H + i] = f2bf(hrow[i] + 0.5f * s);
    }
}

// ---------- kernels 4/5: bf16 GEMM, f32 accumulate via v_wmma_f32_16x16x32_bf16 ----------
// C[M,N] = act(A[MxK] * BT[NxK]^T + bias). WG tile 128x128, 8 waves, wave tile 32x64.
// Tiles staged with global_load_async_to_lds_b128 (ASYNCcnt), double-buffered.
template<bool GELU, bool OUT_BF16>
__global__ __launch_bounds__(256) void k_gemm_bf16(
    const unsigned short* __restrict__ A,    // [M x K] bf16 row-major
    const unsigned short* __restrict__ BT,   // [N x K] bf16 row-major (B transposed)
    const float* __restrict__ bias,          // [N]
    void* __restrict__ outp, int M, int N, int K) {
    constexpr int LDT = 40;                  // padded LDS row stride in halves (32 + 8)
    __shared__ __align__(16) unsigned short As[2][128 * LDT];
    __shared__ __align__(16) unsigned short Bs[2][128 * LDT];

    const int tid  = threadIdx.x;
    const int wave = tid >> 5, lane = tid & 31;
    const int wm   = wave & 3;               // 4 waves along M (32 rows each)
    const int wn   = wave >> 2;              // 2 waves along N (64 cols each)
    const int hsel = lane >> 4, cidx = lane & 15;
    const long blockM = (long)blockIdx.y * 128;
    const long blockN = (long)blockIdx.x * 128;

    const int lrow = tid >> 1, lpart = tid & 1;  // 2 threads per tile row, 32B each

    // Issue async global->LDS copies for one K-step tile (A and B), no VGPR staging.
    auto stage = [&](int buf, int kt) {
        const unsigned short* gA = A  + (size_t)(blockM + lrow) * K + (size_t)kt * 32 + lpart * 16;
        const unsigned short* gB = BT + (size_t)(blockN + lrow) * K + (size_t)kt * 32 + lpart * 16;
        const unsigned la = (unsigned)(uintptr_t)(As[buf] + lrow * LDT + lpart * 16);
        const unsigned lb = (unsigned)(uintptr_t)(Bs[buf] + lrow * LDT + lpart * 16);
        async_copy32(la, gA);
        async_copy32(lb, gB);
    };

    // Gather one 16x32 bf16 fragment from padded LDS tile.
    // ISA layout (16-bit A/B 16x32): lanes 0-15 row r hold K {0..7,16..23},
    // lanes 16-31 hold K {8..15,24..31} -> two ds_load_b128 per lane.
    auto load_frag = [&](const unsigned short* base, int row0) -> bf16x16 {
        const unsigned short* p = base + (row0 + cidx) * LDT + hsel * 8;
        union { bf16x16 v; uint4 q[2]; } u;
        u.q[0] = *(const uint4*)(p);
        u.q[1] = *(const uint4*)(p + 16);
        return u.v;
    };

    f32x8 acc[2][4];
#pragma unroll
    for (int mi = 0; mi < 2; ++mi)
#pragma unroll
        for (int ni = 0; ni < 4; ++ni)
            acc[mi][ni] = f32x8{0.f, 0.f, 0.f, 0.f, 0.f, 0.f, 0.f, 0.f};

    const int KT = K >> 5;
    stage(0, 0);
    wait_async0();
    __syncthreads();

    for (int kt = 0; kt < KT; ++kt) {
        const int buf = kt & 1;
        if (kt + 1 < KT) stage(buf ^ 1, kt + 1);   // async fill of other buffer
        if (kt + 2 < KT) {  // L2 prefetch of tile after next (global_prefetch_b8)
            __builtin_prefetch(A  + (size_t)(blockM + lrow) * K + (size_t)(kt + 2) * 32, 0, 1);
            __builtin_prefetch(BT + (size_t)(blockN + lrow) * K + (size_t)(kt + 2) * 32, 0, 1);
        }

        bf16x16 af0 = load_frag(As[buf], wm * 32);
        bf16x16 af1 = load_frag(As[buf], wm * 32 + 16);
        bf16x16 bf0 = load_frag(Bs[buf], wn * 64);
        bf16x16 bf1 = load_frag(Bs[buf], wn * 64 + 16);
        bf16x16 bf2 = load_frag(Bs[buf], wn * 64 + 32);
        bf16x16 bf3 = load_frag(Bs[buf], wn * 64 + 48);

        acc[0][0] = __builtin_amdgcn_wmma_f32_16x16x32_bf16(false, af0, false, bf0, (short)0, acc[0][0], false, false);
        acc[0][1] = __builtin_amdgcn_wmma_f32_16x16x32_bf16(false, af0, false, bf1, (short)0, acc[0][1], false, false);
        acc[0][2] = __builtin_amdgcn_wmma_f32_16x16x32_bf16(false, af0, false, bf2, (short)0, acc[0][2], false, false);
        acc[0][3] = __builtin_amdgcn_wmma_f32_16x16x32_bf16(false, af0, false, bf3, (short)0, acc[0][3], false, false);
        acc[1][0] = __builtin_amdgcn_wmma_f32_16x16x32_bf16(false, af1, false, bf0, (short)0, acc[1][0], false, false);
        acc[1][1] = __builtin_amdgcn_wmma_f32_16x16x32_bf16(false, af1, false, bf1, (short)0, acc[1][1], false, false);
        acc[1][2] = __builtin_amdgcn_wmma_f32_16x16x32_bf16(false, af1, false, bf2, (short)0, acc[1][2], false, false);
        acc[1][3] = __builtin_amdgcn_wmma_f32_16x16x32_bf16(false, af1, false, bf3, (short)0, acc[1][3], false, false);

        wait_async0();          // async tile for buf^1 fully in LDS
        __syncthreads();        // make it visible workgroup-wide
    }

    // Epilogue. C/D layout: VGPR r -> M = r + 8*(lane>=16), N = lane&15.
#pragma unroll
    for (int mi = 0; mi < 2; ++mi) {
#pragma unroll
        for (int ni = 0; ni < 4; ++ni) {
            const long gm0 = blockM + wm * 32 + mi * 16 + hsel * 8;
            const long gn  = blockN + wn * 64 + ni * 16 + cidx;
            const float bv = bias[gn];
#pragma unroll
            for (int r = 0; r < 8; ++r) {
                float v = acc[mi][ni][r] + bv;
                if (GELU) v = gelu_exact(v);
                if (OUT_BF16) ((unsigned short*)outp)[(gm0 + r) * N + gn] = f2bf(v);
                else          ((float*)outp)[(gm0 + r) * N + gn] = v;
            }
        }
    }
}

// ---------- launch ----------
extern "C" void kernel_launch(void* const* d_in, const int* in_sizes, int n_in,
                              void* d_out, int out_size, void* d_ws, size_t ws_size,
                              hipStream_t stream) {
    const float* hsd = (const float*)d_in[0];   // [2,2048,2048]
    const float* Pe  = (const float*)d_in[1];   // [2048,16]
    const float* Pr  = (const float*)d_in[2];   // [16,2048]
    const float* Adj = (const float*)d_in[3];   // [16,16]
    const float* W1  = (const float*)d_in[4];   // [16,17,8]
    const float* b1  = (const float*)d_in[5];   // [16,8]
    const float* W2  = (const float*)d_in[6];   // [16,8]
    const float* b2  = (const float*)d_in[7];   // [16]
    const float* fw1 = (const float*)d_in[8];   // [2048,8192]
    const float* fb1 = (const float*)d_in[9];   // [8192]
    const float* fw2 = (const float*)d_in[10];  // [8192,2048]
    const float* fb2 = (const float*)d_in[11];  // [2048]

    const long M = 4096, H = 2048, F = 8192;

    unsigned short* modified = (unsigned short*)d_ws;   // [M x H] bf16   (16 MiB)
    unsigned short* w1T      = modified + M * H;        // [F x H] bf16   (32 MiB)
    unsigned short* h1       = w1T + F * H;             // [M x F] bf16   (64 MiB)
    unsigned short* w2T      = h1 + M * F;              // [H x F] bf16   (32 MiB)

    // 1-2) convert+transpose FFN weights to bf16 [N x K]
    k_transpose_f32_bf16<<<dim3(F / 32, H / 32), dim3(32, 8), 0, stream>>>(fw1, w1T, (int)H, (int)F);
    k_transpose_f32_bf16<<<dim3(H / 32, F / 32), dim3(32, 8), 0, stream>>>(fw2, w2T, (int)F, (int)H);

    // 3) causal variables + mechanisms + routing -> modified (bf16)
    k_causal_modify<<<dim3(512), dim3(256), 0, stream>>>(hsd, Pe, Pr, Adj, W1, b1, W2, b2, modified);

    // 4) ffn_h = gelu(modified @ ffn_w1 + b1) -> bf16
    k_gemm_bf16<true, true><<<dim3(F / 128, M / 128), dim3(256), 0, stream>>>(
        modified, w1T, fb1, (void*)h1, (int)M, (int)F, (int)H);

    // 5) out = ffn_h @ ffn_w2 + b2 -> fp32
    k_gemm_bf16<false, false><<<dim3(H / 128, M / 128), dim3(256), 0, stream>>>(
        h1, w2T, fb2, d_out, (int)M, (int)H, (int)F);
}
